// HebbianConv2d_89781996355860
// MI455X (gfx1250) — matrix operand order repaired and checked by reading, not compiled, not run on hardware
//
#include <hip/hip_runtime.h>

typedef __attribute__((ext_vector_type(16))) __bf16 v16bf;
typedef __attribute__((ext_vector_type(8)))  float  v8f;

#define B_   32
#define CIN  32
#define COUT 64
#define H_   128
#define W_   128
#define HP   132   // padded
#define WP   132

// ---- workspace layout (bytes) ----
// xbf  : NHWC padded bf16 [B][132][132][32]    -> 35,684,352 B
// wpack: WMMA-B fragments bf16 [4][25][32][16] -> 102,400 B
// y    : NHWC f32 [B][128][128][64]            -> 134,217,728 B
#define OFF_XBF   0
#define OFF_WPACK 35684352
#define OFF_Y     35786752

__device__ __forceinline__ unsigned short f2bf(float f) {
    union { float f; unsigned u; } cv; cv.f = f;
    unsigned u = cv.u;
    unsigned r = (u + 0x7FFFu + ((u >> 16) & 1u)) >> 16;   // RNE
    return (unsigned short)r;
}

__device__ __forceinline__ int reflect128(int s) {
    if (s < 0)   s = -1 - s;      // symmetric padding
    if (s > 127) s = 255 - s;
    return s;
}

// ---------------------------------------------------------------------------
// Kernel 1: symmetric pad + f32->bf16, NCHW -> NHWC [B][132][132][32]
// ---------------------------------------------------------------------------
__global__ void pad_convert(const float* __restrict__ x, unsigned short* __restrict__ xbf) {
    int id = blockIdx.x * blockDim.x + threadIdx.x;           // 32*132*132 = 557568
    int b  = id / (HP * WP);
    int r  = id - b * (HP * WP);
    int hp = r / WP;
    int wp = r - hp * WP;
    int hs = reflect128(hp - 2);
    int ws = reflect128(wp - 2);
    const float* src = x + (((long)b * CIN) * H_ + hs) * W_ + ws;
    unsigned pk[CIN / 2];
#pragma unroll
    for (int c = 0; c < CIN / 2; ++c) {
        unsigned short lo = f2bf(src[(2 * c + 0) * (H_ * W_)]);
        unsigned short hi = f2bf(src[(2 * c + 1) * (H_ * W_)]);
        pk[c] = (unsigned)lo | ((unsigned)hi << 16);
    }
    uint4* dst = (uint4*)(xbf + (long)id * CIN);
#pragma unroll
    for (int q = 0; q < CIN / 8; ++q)
        dst[q] = make_uint4(pk[4 * q], pk[4 * q + 1], pk[4 * q + 2], pk[4 * q + 3]);
}

// ---------------------------------------------------------------------------
// Kernel 2: pack weights into WMMA-B fragment order.
// K ordering: K = r*32 + cin,  r = kh*5 + kw  (25 K-blocks of 32).
// B 32x16 bf16 layout: lanes 0-15 hold N=lane, K(elem i) = i<8 ? i : i+8; lanes 16-31 K += 8.
// ---------------------------------------------------------------------------
__global__ void pack_weight(const float* __restrict__ w, unsigned short* __restrict__ wpack) {
    int idx = blockIdx.x * blockDim.x + threadIdx.x;          // 4*25*32*16 = 51200
    int i    = idx & 15;
    int lane = (idx >> 4) & 31;
    int r    = (idx >> 9) % 25;
    int g    = idx / (16 * 32 * 25);
    int cout  = g * 16 + (lane & 15);
    int khalf = (lane >> 4) * 16;
    int cin   = (i < 8) ? ((lane >> 4) * 8 + i) : (16 + (lane >> 4) * 8 + (i - 8));
    int kh = r / 5;
    int kw = r - kh * 5;
    float v = w[(((long)cout * CIN + cin) * 5 + kh) * 5 + kw];
    wpack[idx] = f2bf(v);
}

// ---------------------------------------------------------------------------
// Kernel 3: implicit-GEMM conv via v_wmma_f32_16x16x32_bf16.
// grid = B * H/2 blocks (TWO output rows each), 256 threads = 8 waves.
// Wave wv owns pixels w0 = wv*16 .. +15 (M), both rows, all 4 cout-groups (N).
// LDS: contiguous NHWC slab rows h0..h0+5 : [6][132][32] bf16 = 50,688 B,
// staged with GLOBAL_LOAD_ASYNC_TO_LDS_B128 (ASYNCcnt, no VGPR round-trip).
// A 16-bit layout: lane<16 -> M=lane, K(elem i) = i<8 ? i : i+8  (K 0-7,16-23)
//                  lane>=16 -> M=lane-16, K += 8                  (K 8-15,24-31)
// With K = r*32 + cin this is two 16B LDS loads: cin c0..c0+7 and c0+16..c0+23.
// ---------------------------------------------------------------------------
#define SLAB_Q 3168   // 6*132*32 bf16 = 50688 B = 3168 uint4

__global__ void __launch_bounds__(256) conv_wmma(const unsigned short* __restrict__ xbf,
                                                 const unsigned short* __restrict__ wpack,
                                                 float* __restrict__ y) {
    __shared__ uint4 ldsq[SLAB_Q];
    const unsigned short* lds = (const unsigned short*)ldsq;

    int b  = blockIdx.x >> 6;
    int h0 = (blockIdx.x & 63) << 1;

    // async-stage the 6-row NHWC slab (contiguous in global) into LDS
    const uint4* src = (const uint4*)(xbf + ((long)(b * HP + h0) * WP) * CIN);
    for (int i = threadIdx.x; i < SLAB_Q; i += 256) {
        unsigned           ldsoff = (unsigned)(unsigned long long)(const void*)&ldsq[i];
        unsigned long long gaddr  = (unsigned long long)(const void*)&src[i];
        asm volatile("global_load_async_to_lds_b128 %0, %1, off"
                     :: "v"(ldsoff), "v"(gaddr) : "memory");
    }
    asm volatile("s_wait_asynccnt 0x0" ::: "memory");
    __syncthreads();

    int lane = threadIdx.x & 31;
    int wv   = threadIdx.x >> 5;
    int w0   = wv * 16;
    int M    = lane & 15;
    int hi   = lane >> 4;
    int c0   = hi * 8;

    v8f acc[2][4];
#pragma unroll
    for (int d = 0; d < 2; ++d)
#pragma unroll
        for (int g = 0; g < 4; ++g)
            acc[d][g] = (v8f){0, 0, 0, 0, 0, 0, 0, 0};

    union FragU { v16bf v; uint4 q[2]; };

#pragma unroll 5
    for (int r = 0; r < 25; ++r) {
        int kh = r / 5;
        int kw = r - kh * 5;
        int xw = w0 + M + kw;                         // <= 131
        int ai0 = ((kh + 0) * WP + xw) * CIN + c0;    // bf16 idx, 16B aligned
        int ai1 = ((kh + 1) * WP + xw) * CIN + c0;

        FragU a0, a1;
        a0.q[0] = *(const uint4*)&lds[ai0];
        a0.q[1] = *(const uint4*)&lds[ai0 + 16];
        a1.q[0] = *(const uint4*)&lds[ai1];
        a1.q[1] = *(const uint4*)&lds[ai1 + 16];

        const uint4* wb = (const uint4*)(wpack + ((long)(r * 32 + lane) << 4));
        int rn = (r < 24) ? (r + 1) : 24;             // prefetch next K-step's B frags
        __builtin_prefetch((const void*)(wpack + ((long)(rn * 32 + lane) << 4)), 0, 1);

        // cout-groups are strided by 25*32*16 bf16 = 1600 uint4s
        FragU bb[4];
#pragma unroll
        for (int g = 0; g < 4; ++g) {
            bb[g].q[0] = wb[g * 1600 + 0];
            bb[g].q[1] = wb[g * 1600 + 1];
        }

#pragma unroll
        for (int g = 0; g < 4; ++g) {
            acc[0][g] = __builtin_amdgcn_wmma_f32_16x16x32_bf16(false, a0.v, false, bb[g].v,
                                                                (short)0, acc[0][g], false, false);
            acc[1][g] = __builtin_amdgcn_wmma_f32_16x16x32_bf16(false, a1.v, false, bb[g].v,
                                                                (short)0, acc[1][g], false, false);
        }
    }

    // C/D layout: VGPR j -> M = j + hi*8 (pixel), N = lane&15 (cout within group)
#pragma unroll
    for (int d = 0; d < 2; ++d) {
        float* yrow = y + ((long)(b * H_ + h0 + d) * W_) * COUT;
#pragma unroll
        for (int j = 0; j < 8; ++j) {
            int wp = w0 + j + hi * 8;
            float* p = yrow + (long)wp * COUT + M;
            p[0]  = acc[d][0][j];
            p[16] = acc[d][1][j];
            p[32] = acc[d][2][j];
            p[48] = acc[d][3][j];
        }
    }
}

// ---------------------------------------------------------------------------
// Kernel 4: depthwise 5x5 DoG (SAME, zero pad) + top-1 WTA mask, NHWC -> NCHW
// ---------------------------------------------------------------------------
__global__ void __launch_bounds__(256) dog_wta(const float* __restrict__ y, float* __restrict__ out) {
    int p = blockIdx.x * blockDim.x + threadIdx.x;    // 32*128*128
    int b = p >> 14;
    int rem = p & 16383;
    int h = rem >> 7;
    int w = rem & 127;

    const float inv2se = 1.0f / (2.0f * 1.2f * 1.2f);
    const float inv2si = 1.0f / (2.0f * 1.4f * 1.4f);
    const float ce = 1.0f / (2.0f * 3.14159265358979f * 1.2f * 1.2f);
    const float ci = 1.0f / (2.0f * 3.14159265358979f * 1.4f * 1.4f);
    const float inv_norm = 1.0f / (ce - ci);          // DoG normalized by its center

    float4 acc[16];
#pragma unroll
    for (int q = 0; q < 16; ++q) acc[q] = make_float4(0.f, 0.f, 0.f, 0.f);

    for (int dy = -2; dy <= 2; ++dy) {
        int hh = h + dy;
        if (hh < 0 || hh >= H_) continue;
        for (int dx = -2; dx <= 2; ++dx) {
            int ww = w + dx;
            if (ww < 0 || ww >= W_) continue;
            float r2 = (float)(dx * dx + dy * dy);
            float coef = (ce * __expf(-r2 * inv2se) - ci * __expf(-r2 * inv2si)) * inv_norm;
            const float4* row = (const float4*)(y + ((long)(b * H_ + hh) * W_ + ww) * COUT);
#pragma unroll
            for (int q = 0; q < 16; ++q) {
                float4 v = row[q];
                acc[q].x = fmaf(v.x, coef, acc[q].x);
                acc[q].y = fmaf(v.y, coef, acc[q].y);
                acc[q].z = fmaf(v.z, coef, acc[q].z);
                acc[q].w = fmaf(v.w, coef, acc[q].w);
            }
        }
    }

    // top-1 over channels (softmax is monotone -> compare raw values)
    float mx = -3.4e38f;
#pragma unroll
    for (int q = 0; q < 16; ++q) {
        mx = fmaxf(mx, fmaxf(fmaxf(acc[q].x, acc[q].y), fmaxf(acc[q].z, acc[q].w)));
    }

    long obase = ((long)b * COUT) * (H_ * W_) + (long)h * W_ + w;
#pragma unroll
    for (int q = 0; q < 16; ++q) {
        float v0 = acc[q].x, v1 = acc[q].y, v2 = acc[q].z, v3 = acc[q].w;
        out[obase + (long)(4 * q + 0) * (H_ * W_)] = (v0 >= mx) ? v0 : 0.0f;
        out[obase + (long)(4 * q + 1) * (H_ * W_)] = (v1 >= mx) ? v1 : 0.0f;
        out[obase + (long)(4 * q + 2) * (H_ * W_)] = (v2 >= mx) ? v2 : 0.0f;
        out[obase + (long)(4 * q + 3) * (H_ * W_)] = (v3 >= mx) ? v3 : 0.0f;
    }
}

extern "C" void kernel_launch(void* const* d_in, const int* in_sizes, int n_in,
                              void* d_out, int out_size, void* d_ws, size_t ws_size,
                              hipStream_t stream) {
    const float* x      = (const float*)d_in[0];
    const float* weight = (const float*)d_in[1];
    float* out = (float*)d_out;
    char*  ws  = (char*)d_ws;

    unsigned short* xbf   = (unsigned short*)(ws + OFF_XBF);
    unsigned short* wpack = (unsigned short*)(ws + OFF_WPACK);
    float*          ybuf  = (float*)(ws + OFF_Y);

    pad_convert<<<2178, 256, 0, stream>>>(x, xbf);                   // 32*132*132 threads
    pack_weight<<<200, 256, 0, stream>>>(weight, wpack);             // 51200 threads
    conv_wmma  <<<B_ * (H_ / 2), 256, 0, stream>>>(xbf, wpack, ybuf); // 2048 two-row blocks
    dog_wta    <<<2048, 256, 0, stream>>>(ybuf, out);                // 524288 pixels
}
